// MinimalTriXLayer_65884798321353
// MI455X (gfx1250) — compile-verified, matrix operand order checked by reading, not run on hardware
//
#include <hip/hip_runtime.h>
#include <hip/hip_bf16.h>

typedef __bf16 bf16_t;
typedef __attribute__((ext_vector_type(16))) __bf16 v16bf;
typedef __attribute__((ext_vector_type(8)))  __bf16 v8bf;
typedef __attribute__((ext_vector_type(4)))  __bf16 v4bf;
typedef __attribute__((ext_vector_type(8)))  float  v8f;

#define THRESH 0.3f
#define BM 64
#define BN 128
#define BK 32
#define NTHREADS 512

// ---------------- CDNA5 async global->LDS copy (ASYNCcnt-tracked) ----------
#if defined(__gfx1250__)
#define USE_ASYNC 1
#else
#define USE_ASYNC 0
#endif

__device__ __forceinline__ void copy16(const float* g, float* l) {
#if USE_ASYNC
    // GLOBAL_LOAD_ASYNC_TO_LDS_B128: vdst = LDS byte address (low 32 bits of
    // the generic pointer), vaddr = 64-bit global address, no saddr.
    unsigned int lds = (unsigned int)(unsigned long long)l;
    asm volatile("global_load_async_to_lds_b128 %0, %1, off"
                 :
                 : "v"(lds), "v"((unsigned long long)g)
                 : "memory");
#else
    *(float4*)l = *(const float4*)g;
#endif
}

__device__ __forceinline__ void wait_async() {
#if USE_ASYNC
#if __has_builtin(__builtin_amdgcn_s_wait_asynccnt)
    __builtin_amdgcn_s_wait_asynccnt(0);
#else
    asm volatile("s_wait_asynccnt 0" ::: "memory");
#endif
#endif
}

// convert float4 -> bf16 hi/lo and store (one element converted exactly once)
__device__ __forceinline__ void store_cvt(float4 v, bf16_t* ph, bf16_t* pl) {
    float f[4] = {v.x, v.y, v.z, v.w};
    v4bf h, l;
#pragma unroll
    for (int j = 0; j < 4; ++j) {
        bf16_t hb = (bf16_t)f[j];
        h[j] = hb;
        l[j] = (bf16_t)(f[j] - (float)hb);
    }
    *(v4bf*)ph = h;
    *(v4bf*)pl = l;
}

// build a 16xK bf16 fragment from an LDS row (wave32 K-striping)
__device__ __forceinline__ v16bf frag(const bf16_t* row, int kbase) {
    v8bf x0 = *(const v8bf*)(row + kbase);
    v8bf x1 = *(const v8bf*)(row + 16 + kbase);
    return __builtin_shufflevector(x0, x1,
                                   0,1,2,3,4,5,6,7,8,9,10,11,12,13,14,15);
}

// ---------------- Kernel 0: zero per-expert counters ----------------
__global__ void k_init(int* counts, int T) {
    if ((int)threadIdx.x < T) counts[threadIdx.x] = 0;
}

// ---------------- Kernel 1: routing ----------------
// sigs_raw async-staged to LDS (ASYNCcnt), quantized once, then one wave32
// per row computes all 16 ternary scores and argmax-routes the row.
__global__ void k_route(const float* __restrict__ x,
                        const float* __restrict__ sigs_raw,
                        float* __restrict__ idx_out,   // [B] (as float)
                        int*   __restrict__ counts,    // [T]
                        int*   __restrict__ bucket,    // [T][B]
                        int B, int D) {
    __shared__ float  sraw[16 * 1024];   // 64 KB raw (async landing zone)
    __shared__ bf16_t sq[16 * 1024];     // 32 KB ternary (exact in bf16)
    const int tid = threadIdx.x;
    for (int i = tid * 4; i < 16 * 1024; i += (int)blockDim.x * 4)
        copy16(sigs_raw + i, &sraw[i]);
    wait_async();
    __syncthreads();
    for (int i = tid; i < 16 * 1024; i += blockDim.x) {
        float s = sraw[i];
        float q = (s > THRESH) ? 1.0f : ((s < -THRESH) ? -1.0f : 0.0f);
        sq[i] = (bf16_t)q;
    }
    __syncthreads();

    const int wave = tid >> 5, lane = tid & 31;
    const int row  = blockIdx.x * (blockDim.x >> 5) + wave;
    if (row >= B) return;

    float acc[16];
#pragma unroll
    for (int t = 0; t < 16; ++t) acc[t] = 0.0f;

    const float4* xr = (const float4*)(x + (size_t)row * D);
    for (int d4 = lane; d4 < (D >> 2); d4 += 32) {
        float4 xv = xr[d4];
#pragma unroll
        for (int t = 0; t < 16; ++t) {
            v4bf sv = *(const v4bf*)(sq + t * 1024 + (d4 << 2));
            acc[t] += xv.x * (float)sv[0] + xv.y * (float)sv[1] +
                      xv.z * (float)sv[2] + xv.w * (float)sv[3];
        }
    }
#pragma unroll
    for (int t = 0; t < 16; ++t) {
        float v = acc[t];
        for (int off = 16; off > 0; off >>= 1) v += __shfl_xor(v, off, 32);
        acc[t] = v;
    }
    if (lane == 0) {
        int best = 0; float bv = acc[0];
#pragma unroll
        for (int t = 1; t < 16; ++t) if (acc[t] > bv) { bv = acc[t]; best = t; }
        idx_out[row] = (float)best;
        int pos = atomicAdd(&counts[best], 1);
        bucket[best * B + pos] = row;
    }
}

// ---------------- Kernel 2: grouped expert GEMM ----------------
// 512 threads = 16 waves. Tile: M=64 gathered rows x N=128 classes, K step 32.
// Split-precision bf16 (hi/lo) converted ONCE at stage time into LDS;
// double-buffered, one barrier per K-step, global loads for step i+1
// register-pipelined behind the WMMA section. 6 WMMAs per wave per K-step.
__global__ void __launch_bounds__(NTHREADS)
k_gemm(const float* __restrict__ x,
       const float* __restrict__ W,
       const float* __restrict__ bias,
       const int*   __restrict__ counts,
       const int*   __restrict__ bucket,
       float* __restrict__ out,
       int B, int D, int C) {
    const int t   = blockIdx.z;
    const int cnt = counts[t];
    const int m0  = blockIdx.y * BM;
    if (m0 >= cnt) return;
    const int n0  = blockIdx.x * BN;

    __shared__ int    rowids[BM];
    __shared__ bf16_t Ah[2][BM][BK], Al[2][BM][BK];   // 4x4 KB
    __shared__ bf16_t Bh[2][BN][BK], Bl[2][BN][BK];   // 4x8 KB

    const int tid = threadIdx.x;
    if (tid < BM) {
        int r = m0 + tid;
        rowids[tid] = (r < cnt) ? bucket[t * B + r] : -1;
    }
    __syncthreads();

    // ---- per-thread staging addresses (fixed across K loop) ----
    const int eA = tid * 4;                  // 64x32 f32: one float4/thread
    const int rA = eA >> 5, kA = eA & 31;
    int growA = rowids[rA];
    if (growA < 0) growA = rowids[0];        // valid filler row (never stored)
    const float* gA = x + (size_t)growA * D + kA;

    const int eB0 = tid * 4,        rB0 = eB0 >> 5, kB0 = eB0 & 31;
    const int eB1 = tid * 4 + 2048, rB1 = eB1 >> 5, kB1 = eB1 & 31;
    int cls0 = n0 + rB0; if (cls0 >= C) cls0 = C - 1;   // clamped filler
    int cls1 = n0 + rB1; if (cls1 >= C) cls1 = C - 1;
    const float* gB0 = W + ((size_t)t * C + cls0) * D + kB0;
    const float* gB1 = W + ((size_t)t * C + cls1) * D + kB1;

    const int wave  = tid >> 5, lane = tid & 31;
    const int mt    = wave >> 2;             // 0..3
    const int nt0   = wave & 3;              // 0..3
    const int nt1   = nt0 + 4;               // 4..7
    const int kbase = (lane >= 16) ? 8 : 0;  // wave32 K striping
    const int l15   = lane & 15;
    const int ar    = mt * 16 + l15;
    const int bn0   = nt0 * 16 + l15;
    const int bn1   = nt1 * 16 + l15;

    v8f c0 = {}, c1 = {};

    // prologue: global loads for step 0
    float4 ra  = *(const float4*)gA;
    float4 rb0 = *(const float4*)gB0;
    float4 rb1 = *(const float4*)gB1;

    const int nsteps = D / BK;
    for (int i = 0; i < nsteps; ++i) {
        const int buf = i & 1;
        // convert once and stage bf16 hi/lo tiles
        store_cvt(ra,  &Ah[buf][rA][kA],   &Al[buf][rA][kA]);
        store_cvt(rb0, &Bh[buf][rB0][kB0], &Bl[buf][rB0][kB0]);
        store_cvt(rb1, &Bh[buf][rB1][kB1], &Bl[buf][rB1][kB1]);
        // pipeline: issue next step's global loads before the barrier
        if (i + 1 < nsteps) {
            const int kk = (i + 1) * BK;
            ra  = *(const float4*)(gA  + kk);
            rb0 = *(const float4*)(gB0 + kk);
            rb1 = *(const float4*)(gB1 + kk);
        }
        __syncthreads();

        v16bf a_hi = frag(&Ah[buf][ar][0],  kbase);
        v16bf a_lo = frag(&Al[buf][ar][0],  kbase);
        v16bf b0h  = frag(&Bh[buf][bn0][0], kbase);
        v16bf b0l  = frag(&Bl[buf][bn0][0], kbase);
        v16bf b1h  = frag(&Bh[buf][bn1][0], kbase);
        v16bf b1l  = frag(&Bl[buf][bn1][0], kbase);

        c0 = __builtin_amdgcn_wmma_f32_16x16x32_bf16(false, a_hi, false, b0h, (short)0, c0, false, false);
        c0 = __builtin_amdgcn_wmma_f32_16x16x32_bf16(false, a_hi, false, b0l, (short)0, c0, false, false);
        c0 = __builtin_amdgcn_wmma_f32_16x16x32_bf16(false, a_lo, false, b0h, (short)0, c0, false, false);
        c1 = __builtin_amdgcn_wmma_f32_16x16x32_bf16(false, a_hi, false, b1h, (short)0, c1, false, false);
        c1 = __builtin_amdgcn_wmma_f32_16x16x32_bf16(false, a_hi, false, b1l, (short)0, c1, false, false);
        c1 = __builtin_amdgcn_wmma_f32_16x16x32_bf16(false, a_lo, false, b1h, (short)0, c1, false, false);
        // no trailing barrier: double buffer + next iteration's barrier
    }

    // ---- store: lanes 0-15 -> M=j, lanes 16-31 -> M=8+j; N = lane&15 ----
    const int mbase = (lane >= 16) ? 8 : 0;
    const int cn0   = n0 + nt0 * 16 + l15;
    const int cn1   = n0 + nt1 * 16 + l15;
#pragma unroll
    for (int j = 0; j < 8; ++j) {
        int m    = mt * 16 + mbase + j;
        int grow = rowids[m];
        if (grow >= 0) {
            if (cn0 < C) out[(size_t)grow * C + cn0] = c0[j] + bias[t * C + cn0];
            if (cn1 < C) out[(size_t)grow * C + cn1] = c1[j] + bias[t * C + cn1];
        }
    }
}

// ---------------- launcher ----------------
extern "C" void kernel_launch(void* const* d_in, const int* in_sizes, int n_in,
                              void* d_out, int out_size, void* d_ws, size_t ws_size,
                              hipStream_t stream) {
    const float* x    = (const float*)d_in[0];   // [B, D]
    const float* sigs = (const float*)d_in[1];   // [T, D]
    const float* W    = (const float*)d_in[2];   // [T, C, D]
    const float* bias = (const float*)d_in[3];   // [T, C]
    float* out = (float*)d_out;

    const int D = 1024, T = 16;
    const int B = in_sizes[0] / D;
    const int C = in_sizes[3] / T;

    int* counts = (int*)d_ws;
    int* bucket = counts + 32;                   // [T][B] ints after 128 B
    float* idx_out = out + (size_t)B * C;        // tile_indices tail

    k_init<<<dim3(1), dim3(32), 0, stream>>>(counts, T);

    k_route<<<dim3((B + 7) / 8), dim3(256), 0, stream>>>(x, sigs, idx_out,
                                                         counts, bucket, B, D);

    dim3 g((C + BN - 1) / BN, (B + BM - 1) / BM, T);
    k_gemm<<<g, dim3(NTHREADS), 0, stream>>>(x, W, bias, counts, bucket, out,
                                             B, D, C);
}
